// BrainEncodeEmbed_88742614270020
// MI455X (gfx1250) — compile-verified, matrix operand order checked by reading, not run on hardware
//
#include <hip/hip_runtime.h>
#include <hip/hip_bf16.h>
#include <stdint.h>

// ---------------- problem constants ----------------
#define NN   59392      // nodes
#define FF   116        // input features
#define HH   256        // hidden
#define EE   950272     // edges
#define EDD  5          // edge attr dim
#define EMB  16         // group embedding dim
#define CIN  132        // F + EMB
#define KP   160        // CIN padded to multiple of 32
#define BN_EPS 1e-5f

typedef __attribute__((ext_vector_type(16))) __bf16 v16bf;
typedef __attribute__((ext_vector_type(8)))  __bf16 v8bf;
typedef __attribute__((ext_vector_type(8)))  float  v8f;

// K index inside a 16-bit A/B WMMA fragment: element i of lane-half `half`
// (two contiguous 8-element runs: [half*8, half*8+8) and [16+half*8, 16+half*8+8))
__device__ __forceinline__ int kmap(int half, int i) {
    return (i < 8) ? (half * 8 + i) : (16 + half * 8 + (i - 8));
}

// Assemble a 16-bit WMMA fragment from two 8-element runs at a base pointer
// with a compile-time element offset (folds into immediate-offset b128 loads).
__device__ __forceinline__ v16bf load_frag_ofs(const __bf16* __restrict__ base,
                                               int elem_ofs) {
    const v8bf lo = *(const v8bf*)(base + elem_ofs);
    const v8bf hi = *(const v8bf*)(base + elem_ofs + 16);
    v16bf b;
#pragma unroll
    for (int i = 0; i < 8; ++i) { b[i] = lo[i]; b[i + 8] = hi[i]; }
    return b;
}

__device__ __forceinline__ float atomic_fadd_agent(float* p, float v) {
    return __hip_atomic_fetch_add(p, v, __ATOMIC_RELAXED, __HIP_MEMORY_SCOPE_AGENT);
}

// ---------------- prep: bf16 transposed weights + zero BN stats ----------------
__global__ void prep_kernel(const float* __restrict__ W_in,
                            const float* __restrict__ W1,
                            const float* __restrict__ W2,
                            __bf16* __restrict__ wtin,   // [HH][KP]
                            __bf16* __restrict__ w1t,    // [HH][HH]
                            __bf16* __restrict__ w2t,    // [HH][HH]
                            float* __restrict__ stats) { // [512]
    int t = blockIdx.x * blockDim.x + threadIdx.x;
    int stride = gridDim.x * blockDim.x;
    for (int idx = t; idx < HH * KP; idx += stride) {
        int n = idx / KP, k = idx % KP;
        wtin[idx] = (__bf16)((k < CIN) ? W_in[(size_t)k * HH + n] : 0.0f);
    }
    for (int idx = t; idx < HH * HH; idx += stride) {
        int n = idx >> 8, k = idx & 255;
        w1t[idx] = (__bf16)W1[(size_t)k * HH + n];
        w2t[idx] = (__bf16)W2[(size_t)k * HH + n];
    }
    if (t < 512) stats[t] = 0.0f;
}

// ---------------- GEMM1: h = ReLU([x|emb] @ W_in + b_in) ----------------
// 256 threads = 8 waves per block. Each wave owns a 16-row tile and HALF of the
// 16 column tiles (8 tiles = 64 acc VGPRs -> no spills). Two waves per m-tile.
__global__ void gemm1_kernel(const float* __restrict__ x,
                             const long long* __restrict__ gids,
                             const float* __restrict__ gemb,
                             const __bf16* __restrict__ wtin,
                             const float* __restrict__ b_in,
                             float* __restrict__ h) {
    const int wave  = threadIdx.x >> 5;
    const int lane  = threadIdx.x & 31;
    const int half  = lane >> 4;
    const int m     = lane & 15;
    const int mtile = blockIdx.x * 4 + (wave >> 1);
    const int nt0   = (wave & 1) * 8;           // first column tile for this wave
    const int row0  = mtile * 16;
    const int rowA  = row0 + m;                 // A-matrix row for this lane
    const int gid   = (int)gids[rowA];
    const float* xrow = x + (size_t)rowA * FF;  // 16B-aligned (116*4 = 29*16)
    const float* erow = gemb + (size_t)gid * EMB;
    // per-wave B base: column (nt0*16 + m); nt adds constant element offsets
    const __bf16* wb = wtin + (size_t)(nt0 * 16 + m) * KP;

    v8f acc[8];
#pragma unroll
    for (int nt = 0; nt < 8; ++nt)
#pragma unroll
        for (int r = 0; r < 8; ++r) acc[nt][r] = 0.0f;

#pragma unroll
    for (int kc = 0; kc < KP; kc += 32) {
        v16bf a;
        if (kc + 32 <= FF) {                    // kc = 0,32,64: pure-x, vector loads
            const float* p0 = xrow + kc + half * 8;
            const float* p1 = xrow + kc + 16 + half * 8;
            float4 f0 = *(const float4*)(p0);
            float4 f1 = *(const float4*)(p0 + 4);
            float4 f2 = *(const float4*)(p1);
            float4 f3 = *(const float4*)(p1 + 4);
            a[0]  = (__bf16)f0.x; a[1]  = (__bf16)f0.y;
            a[2]  = (__bf16)f0.z; a[3]  = (__bf16)f0.w;
            a[4]  = (__bf16)f1.x; a[5]  = (__bf16)f1.y;
            a[6]  = (__bf16)f1.z; a[7]  = (__bf16)f1.w;
            a[8]  = (__bf16)f2.x; a[9]  = (__bf16)f2.y;
            a[10] = (__bf16)f2.z; a[11] = (__bf16)f2.w;
            a[12] = (__bf16)f3.x; a[13] = (__bf16)f3.y;
            a[14] = (__bf16)f3.z; a[15] = (__bf16)f3.w;
        } else {                                // boundary chunks: concat + zero pad
#pragma unroll
            for (int i = 0; i < 16; ++i) {
                int k = kc + kmap(half, i);
                float v = 0.0f;
                if (k < FF)        v = xrow[k];
                else if (k < CIN)  v = erow[k - FF];
                a[i] = (__bf16)v;
            }
        }
        const __bf16* wbk = wb + kc + half * 8;
#pragma unroll
        for (int nt = 0; nt < 8; ++nt) {
            v16bf b = load_frag_ofs(wbk, nt * (16 * KP));
            acc[nt] = __builtin_amdgcn_wmma_f32_16x16x32_bf16(
                false, a, false, b, (short)0, acc[nt], false, false);
        }
    }
    // epilogue: bias + ReLU
#pragma unroll
    for (int nt = 0; nt < 8; ++nt) {
        int col = (nt0 + nt) * 16 + m;
        float bias = b_in[col];
#pragma unroll
        for (int r = 0; r < 8; ++r) {
            int row = row0 + half * 8 + r;
            float v = acc[nt][r] + bias;
            h[(size_t)row * HH + col] = fmaxf(v, 0.0f);
        }
    }
}

// ---------------- BN stats: per-column sum / sumsq ----------------
__global__ void col_stats_kernel(const float* __restrict__ h,
                                 float* __restrict__ stats) {
    int c = threadIdx.x;                 // column 0..255
    int r0 = blockIdx.x * 256;
    float s = 0.0f, sq = 0.0f;
    for (int r = r0; r < r0 + 256; ++r) {
        float v = h[(size_t)r * HH + c];
        s += v; sq += v * v;
    }
    atomic_fadd_agent(&stats[c], s);
    atomic_fadd_agent(&stats[HH + c], sq);
}

// ---------------- BN apply (in place) ----------------
__global__ void bn_apply_kernel(float* __restrict__ h,
                                const float* __restrict__ stats,
                                const float* __restrict__ gamma,
                                const float* __restrict__ beta) {
    int idx = blockIdx.x * 256 + threadIdx.x;
    if (idx >= NN * HH) return;
    int c = idx & 255;
    const float inv_n = 1.0f / (float)NN;
    float mu  = stats[c] * inv_n;
    float var = stats[HH + c] * inv_n - mu * mu;
    float s   = rsqrtf(var + BN_EPS) * gamma[c];
    h[idx] = (h[idx] - mu) * s + beta[c];
}

// ---------------- GINE edge pass: msg = ReLU(h[src] + ea@We + be); agg[dst] += msg --------
// 64 threads per edge, 4 channels (float4) per thread. h/agg are L2-resident (61MB each).
__global__ void edge_kernel(const long long* __restrict__ ei,
                            const float* __restrict__ ea,
                            const float* __restrict__ We,
                            const float* __restrict__ be,
                            const float* __restrict__ h,
                            float* __restrict__ agg) {
    long long gidx = (long long)blockIdx.x * 256 + threadIdx.x;
    int e = (int)(gidx >> 6);
    if (e >= EE) return;
    int c0 = (int)(gidx & 63) * 4;

    int src = (int)ei[e];
    int dst = (int)ei[(size_t)EE + e];

    float4 v = *(const float4*)(be + c0);
#pragma unroll
    for (int d = 0; d < EDD; ++d) {
        float a = ea[(size_t)e * EDD + d];
        float4 w = *(const float4*)(We + (size_t)d * HH + c0);
        v.x += a * w.x; v.y += a * w.y; v.z += a * w.z; v.w += a * w.w;
    }
    float4 hs = *(const float4*)(h + (size_t)src * HH + c0);
    float m0 = fmaxf(hs.x + v.x, 0.0f);
    float m1 = fmaxf(hs.y + v.y, 0.0f);
    float m2 = fmaxf(hs.z + v.z, 0.0f);
    float m3 = fmaxf(hs.w + v.w, 0.0f);
    float* base = agg + (size_t)dst * HH + c0;
    atomic_fadd_agent(base + 0, m0);
    atomic_fadd_agent(base + 1, m1);
    atomic_fadd_agent(base + 2, m2);
    atomic_fadd_agent(base + 3, m3);
}

// ---------------- GEMM2: out = ReLU(ReLU((h+agg)@W1+b1)@W2+b2) ----------------
// 8 waves per block; two waves share one 16-row tile (one per 128-column half),
// cooperating through a bf16 LDS staging buffer between the two WMMA GEMMs.
__global__ void gemm2_kernel(const float* __restrict__ h,
                             const float* __restrict__ agg,
                             const __bf16* __restrict__ w1t,
                             const float* __restrict__ b1,
                             const __bf16* __restrict__ w2t,
                             const float* __restrict__ b2,
                             float* __restrict__ out) {
    __shared__ __bf16 tls[4][16][HH];           // 32 KB
    const int wave  = threadIdx.x >> 5;
    const int lane  = threadIdx.x & 31;
    const int half  = lane >> 4;
    const int m     = lane & 15;
    const int mt    = wave >> 1;                // m-tile slot within block
    const int nt0   = (wave & 1) * 8;           // column-tile base for this wave
    const int mtile = blockIdx.x * 4 + mt;
    const int row0  = mtile * 16;
    const int rowA  = row0 + m;

    const float*  hrow = h   + (size_t)rowA * HH;
    const float*  arow = agg + (size_t)rowA * HH;
    const __bf16* wb1  = w1t + (size_t)(nt0 * 16 + m) * HH;
    const __bf16* wb2  = w2t + (size_t)(nt0 * 16 + m) * HH;

    v8f acc[8];
#pragma unroll
    for (int nt = 0; nt < 8; ++nt)
#pragma unroll
        for (int r = 0; r < 8; ++r) acc[nt][r] = 0.0f;

    // ---- GEMM A: z @ W1, z = h + agg computed on the fly ----
    for (int kc = 0; kc < HH; kc += 32) {
        const float* hz = hrow + kc + half * 8;
        const float* az = arow + kc + half * 8;
        v16bf a;
        {
            float4 f0 = *(const float4*)(hz);
            float4 f1 = *(const float4*)(hz + 4);
            float4 g0 = *(const float4*)(az);
            float4 g1 = *(const float4*)(az + 4);
            a[0] = (__bf16)(f0.x + g0.x); a[1] = (__bf16)(f0.y + g0.y);
            a[2] = (__bf16)(f0.z + g0.z); a[3] = (__bf16)(f0.w + g0.w);
            a[4] = (__bf16)(f1.x + g1.x); a[5] = (__bf16)(f1.y + g1.y);
            a[6] = (__bf16)(f1.z + g1.z); a[7] = (__bf16)(f1.w + g1.w);
            float4 f2 = *(const float4*)(hz + 16);
            float4 f3 = *(const float4*)(hz + 20);
            float4 g2 = *(const float4*)(az + 16);
            float4 g3 = *(const float4*)(az + 20);
            a[8]  = (__bf16)(f2.x + g2.x); a[9]  = (__bf16)(f2.y + g2.y);
            a[10] = (__bf16)(f2.z + g2.z); a[11] = (__bf16)(f2.w + g2.w);
            a[12] = (__bf16)(f3.x + g3.x); a[13] = (__bf16)(f3.y + g3.y);
            a[14] = (__bf16)(f3.z + g3.z); a[15] = (__bf16)(f3.w + g3.w);
        }
        const __bf16* wbk = wb1 + kc + half * 8;
#pragma unroll
        for (int nt = 0; nt < 8; ++nt) {
            v16bf b = load_frag_ofs(wbk, nt * (16 * HH));
            acc[nt] = __builtin_amdgcn_wmma_f32_16x16x32_bf16(
                false, a, false, b, (short)0, acc[nt], false, false);
        }
    }
    // bias + ReLU -> LDS (bf16), laid out row-major [16][256]
#pragma unroll
    for (int nt = 0; nt < 8; ++nt) {
        int col = (nt0 + nt) * 16 + m;
        float bias = b1[col];
#pragma unroll
        for (int r = 0; r < 8; ++r) {
            float v = fmaxf(acc[nt][r] + bias, 0.0f);
            tls[mt][half * 8 + r][col] = (__bf16)v;
        }
    }
    __syncthreads();

    // ---- GEMM B: t @ W2, A fragments from LDS (full K), same column half ----
#pragma unroll
    for (int nt = 0; nt < 8; ++nt)
#pragma unroll
        for (int r = 0; r < 8; ++r) acc[nt][r] = 0.0f;

    for (int kc = 0; kc < HH; kc += 32) {
        const v8bf lo = *(const v8bf*)&tls[mt][m][kc + half * 8];
        const v8bf hi = *(const v8bf*)&tls[mt][m][kc + 16 + half * 8];
        v16bf a;
#pragma unroll
        for (int i = 0; i < 8; ++i) { a[i] = lo[i]; a[i + 8] = hi[i]; }
        const __bf16* wbk = wb2 + kc + half * 8;
#pragma unroll
        for (int nt = 0; nt < 8; ++nt) {
            v16bf b = load_frag_ofs(wbk, nt * (16 * HH));
            acc[nt] = __builtin_amdgcn_wmma_f32_16x16x32_bf16(
                false, a, false, b, (short)0, acc[nt], false, false);
        }
    }
    // bias + ReLU -> out
#pragma unroll
    for (int nt = 0; nt < 8; ++nt) {
        int col = (nt0 + nt) * 16 + m;
        float bias = b2[col];
#pragma unroll
        for (int r = 0; r < 8; ++r) {
            int row = row0 + half * 8 + r;
            float v = fmaxf(acc[nt][r] + bias, 0.0f);
            out[(size_t)row * HH + col] = v;
        }
    }
}

// ---------------- launch ----------------
extern "C" void kernel_launch(void* const* d_in, const int* in_sizes, int n_in,
                              void* d_out, int out_size, void* d_ws, size_t ws_size,
                              hipStream_t stream) {
    const float*     x     = (const float*)d_in[0];
    const long long* ei    = (const long long*)d_in[1];
    const float*     ea    = (const float*)d_in[2];
    const long long* gids  = (const long long*)d_in[3];
    const float*     gemb  = (const float*)d_in[4];
    const float*     W_in  = (const float*)d_in[5];
    const float*     b_in  = (const float*)d_in[6];
    const float*     gamma = (const float*)d_in[7];
    const float*     beta  = (const float*)d_in[8];
    const float*     We    = (const float*)d_in[9];
    const float*     be    = (const float*)d_in[10];
    const float*     W1    = (const float*)d_in[11];
    const float*     b1    = (const float*)d_in[12];
    const float*     W2    = (const float*)d_in[13];
    const float*     b2    = (const float*)d_in[14];

    float* out = (float*)d_out;

    // workspace layout
    char* ws = (char*)d_ws;
    const size_t HB = (size_t)NN * HH * sizeof(float);     // 60,817,408 B
    float*  h     = (float*)(ws);
    float*  agg   = (float*)(ws + HB);
    float*  stats = (float*)(ws + 2 * HB);                 // 512 floats
    __bf16* wtin  = (__bf16*)(ws + 2 * HB + 2048);         // HH*KP bf16
    __bf16* w1t   = (__bf16*)((char*)wtin + (size_t)HH * KP * 2);
    __bf16* w2t   = (__bf16*)((char*)w1t + (size_t)HH * HH * 2);

    hipMemsetAsync(agg, 0, HB, stream);
    prep_kernel<<<256, 256, 0, stream>>>(W_in, W1, W2, wtin, w1t, w2t, stats);

    gemm1_kernel<<<NN / 64, 256, 0, stream>>>(x, gids, gemb, wtin, b_in, h);
    col_stats_kernel<<<NN / 256, 256, 0, stream>>>(h, stats);
    bn_apply_kernel<<<(NN * HH) / 256, 256, 0, stream>>>(h, stats, gamma, beta);

    edge_kernel<<<(int)(((long long)EE * 64) / 256), 256, 0, stream>>>(
        ei, ea, We, be, h, agg);

    gemm2_kernel<<<NN / 64, 256, 0, stream>>>(h, agg, w1t, b1, w2t, b2, out);

    // tuple output: (out, edge_attr) — pass edge_attr through
    hipMemcpyAsync(out + (size_t)NN * HH, ea, (size_t)EE * EDD * sizeof(float),
                   hipMemcpyDeviceToDevice, stream);
}
